// MHA_27943057227970
// MI455X (gfx1250) — matrix operand order repaired and checked by reading, compiled, not run on hardware
//
#include <hip/hip_runtime.h>
#include <hip/hip_bf16.h>

typedef __attribute__((ext_vector_type(16))) _Float16 v16h;
typedef __attribute__((ext_vector_type(8)))  _Float16 v8h;
typedef __attribute__((ext_vector_type(8)))  float    v8f;
typedef int i32x4 __attribute__((vector_size(16)));

#define GLB_AS __attribute__((address_space(1)))
#define LDS_AS __attribute__((address_space(3)))

#define D_MODEL 1024
#define NHEAD   16
#define DHEAD   64
#define SEQ     2048
#define BATCH   2
#define NTOK    (BATCH * SEQ)   // 4096

#if __has_builtin(__builtin_amdgcn_global_load_async_to_lds_b128)
#define HAVE_ASYNC 1
#else
#define HAVE_ASYNC 0
#endif

static __device__ __forceinline__ v16h cat8(v8h lo, v8h hi) {
    return __builtin_shufflevector(lo, hi, 0,1,2,3,4,5,6,7,8,9,10,11,12,13,14,15);
}
static __device__ __forceinline__ v8f wmma16(v16h a, v16h b, v8f c) {
    // D = A(16x32 f16) x B(32x16 f16) + C(16x16 f32)
    return __builtin_amdgcn_wmma_f32_16x16x32_f16(false, a, false, b, (short)0, c, false, false);
}

// 16-byte global -> LDS copy. Async DMA path on gfx1250 when available.
static __device__ __forceinline__ void cp16(const _Float16* g, _Float16* l) {
#if HAVE_ASYNC
    __builtin_amdgcn_global_load_async_to_lds_b128(
        (GLB_AS i32x4*)g,
        (LDS_AS i32x4*)l, 0, 0);
#else
    *(uint4*)l = *(const uint4*)g;
#endif
}
static __device__ __forceinline__ void async_fence() {
#if HAVE_ASYNC
#if __has_builtin(__builtin_amdgcn_s_wait_asynccnt)
    __builtin_amdgcn_s_wait_asynccnt(0);
#else
    asm volatile("s_wait_asynccnt 0x0" ::: "memory");
#endif
#endif
}

// ---------------------------------------------------------------------------
// Kernel 1: f32 -> f16 elementwise convert (for x)
// ---------------------------------------------------------------------------
__global__ __launch_bounds__(256)
void cvt_f32_to_f16(const float* __restrict__ in, _Float16* __restrict__ out, int n) {
    int i = blockIdx.x * blockDim.x + threadIdx.x;
    int stride = gridDim.x * blockDim.x;
    for (; i < n; i += stride) out[i] = (_Float16)in[i];
}

// ---------------------------------------------------------------------------
// Kernel 2: transpose + convert weight: WT[n][k] = (f16)W[k][n]  (1024x1024)
// ---------------------------------------------------------------------------
__global__ __launch_bounds__(256)
void w_transpose_f16(const float* __restrict__ W, _Float16* __restrict__ WT) {
    __shared__ float t[32][33];
    const int tx = threadIdx.x & 31;
    const int ty = threadIdx.x >> 5;           // 0..7
    const int bx = blockIdx.x, by = blockIdx.y;
#pragma unroll
    for (int i = 0; i < 4; ++i) {
        int k = by * 32 + ty + i * 8;
        t[ty + i * 8][tx] = W[(size_t)k * D_MODEL + bx * 32 + tx];
    }
    __syncthreads();
#pragma unroll
    for (int i = 0; i < 4; ++i) {
        int n = bx * 32 + ty + i * 8;
        WT[(size_t)n * D_MODEL + by * 32 + tx] = (_Float16)t[tx][ty + i * 8];
    }
}

// ---------------------------------------------------------------------------
// Shared GEMM tile machinery: block tile 128(M) x 128(N), K-tile 64,
// double-buffered LDS, one barrier per K-iteration, async prefetch.
// 8 waves as 2(M) x 4(N); each wave 64x32 = 4x2 fragments, 16 WMMA / K-iter.
// LDS pitch 72 f16 (144B): 16B aligned, conflict-free for b128 reads.
// ---------------------------------------------------------------------------
#define KT     64
#define PITCH  72

// stage a 128-row x 64-col f16 tile (row stride = D_MODEL) into LDS
static __device__ __forceinline__ void stage_tile(const _Float16* __restrict__ g,
                                                  _Float16* __restrict__ s, int tid) {
    const int r = tid >> 1;
    const int c = (tid & 1) * 32;
    const _Float16* gp = g + (size_t)r * D_MODEL + c;
    _Float16* sp = s + r * PITCH + c;
#pragma unroll
    for (int i = 0; i < 4; ++i) cp16(gp + i * 8, sp + i * 8);
}

// ---------------------------------------------------------------------------
// Kernel 3: fused QKV projection GEMM.
//   A = x_f16 [4096 x 1024]; B = WT [1024(n) x 1024(k)] selected by blockIdx.z
//   Q,K -> [B,H,S,DH]; V -> transposed [B,H,DH,S]
// ---------------------------------------------------------------------------
__global__ __launch_bounds__(256)
void qkv_gemm(const _Float16* __restrict__ xf, const _Float16* __restrict__ WTbase,
              const float* __restrict__ bq, const float* __restrict__ bk,
              const float* __restrict__ bv,
              _Float16* __restrict__ Qo, _Float16* __restrict__ Ko,
              _Float16* __restrict__ VTo) {
    __shared__ _Float16 sA[2][128 * PITCH];
    __shared__ _Float16 sB[2][128 * PITCH];

    const int tid  = threadIdx.x;
    const int w    = tid >> 5, lane = tid & 31;
    const int ln   = lane & 15, hf = lane >> 4;
    const int wm   = w >> 2,  wn = w & 3;
    const int mb   = blockIdx.y * 128;
    const int nb   = blockIdx.x * 128;
    const int z    = blockIdx.z;   // 0:Q 1:K 2:V

    const _Float16* WT   = WTbase + (size_t)z * D_MODEL * D_MODEL;
    const float*    bias = (z == 0) ? bq : (z == 1) ? bk : bv;

    v8f acc[4][2];
#pragma unroll
    for (int i = 0; i < 4; ++i)
#pragma unroll
        for (int j = 0; j < 2; ++j) acc[i][j] = v8f{};

    const _Float16* gA = xf + (size_t)mb * D_MODEL;
    const _Float16* gB = WT + (size_t)nb * D_MODEL;

    stage_tile(gA, sA[0], tid);
    stage_tile(gB, sB[0], tid);

    const int NIT = D_MODEL / KT;   // 16
    for (int it = 0; it < NIT; ++it) {
        const int buf = it & 1;
        async_fence();
        __syncthreads();
        if (it + 1 < NIT) {
            stage_tile(gA + (it + 1) * KT, sA[buf ^ 1], tid);
            stage_tile(gB + (it + 1) * KT, sB[buf ^ 1], tid);
        }
#pragma unroll
        for (int ks = 0; ks < 2; ++ks) {
            v16h afr[4];
#pragma unroll
            for (int mt = 0; mt < 4; ++mt) {
                const _Float16* p = sA[buf] + (wm * 64 + mt * 16 + ln) * PITCH + ks * 32;
                afr[mt] = cat8(*(const v8h*)(p + 8 * hf), *(const v8h*)(p + 16 + 8 * hf));
            }
#pragma unroll
            for (int nt = 0; nt < 2; ++nt) {
                const _Float16* p = sB[buf] + (wn * 32 + nt * 16 + ln) * PITCH + ks * 32;
                v16h bfr = cat8(*(const v8h*)(p + 8 * hf), *(const v8h*)(p + 16 + 8 * hf));
#pragma unroll
                for (int mt = 0; mt < 4; ++mt)
                    acc[mt][nt] = wmma16(afr[mt], bfr, acc[mt][nt]);
            }
        }
    }

    // epilogue (branch hoisted out of the store loops)
    if (z == 2) {
#pragma unroll
        for (int mt = 0; mt < 4; ++mt)
#pragma unroll
            for (int nt = 0; nt < 2; ++nt) {
                const int n = nb + wn * 32 + nt * 16 + ln;
                const float bn = bias[n];
                const int h = n >> 6, d = n & 63;
#pragma unroll
                for (int v = 0; v < 8; ++v) {
                    const int m = mb + wm * 64 + mt * 16 + v + 8 * hf;
                    const int b = m >> 11, s = m & 2047;
                    VTo[(((size_t)(b * NHEAD + h) * DHEAD + d) * SEQ) + s] =
                        (_Float16)(acc[mt][nt][v] + bn);
                }
            }
    } else {
        _Float16* dst = (z == 0) ? Qo : Ko;
#pragma unroll
        for (int mt = 0; mt < 4; ++mt)
#pragma unroll
            for (int nt = 0; nt < 2; ++nt) {
                const int n = nb + wn * 32 + nt * 16 + ln;
                const float bn = bias[n];
                const int h = n >> 6, d = n & 63;
#pragma unroll
                for (int v = 0; v < 8; ++v) {
                    const int m = mb + wm * 64 + mt * 16 + v + 8 * hf;
                    const int b = m >> 11, s = m & 2047;
                    dst[(((size_t)(b * NHEAD + h) * SEQ + s) * DHEAD) + d] =
                        (_Float16)(acc[mt][nt][v] + bn);
                }
            }
    }
}

// ---------------------------------------------------------------------------
// Kernel 4: flash attention. Block = 128 q rows of one (b,h); 8 waves x 16 rows.
// Q,K: [B,H,S,DH] f16; VT: [B,H,DH,S] f16; AO: [NTOK, D] f16.
// Double-buffered K/V tiles, async prefetch, one barrier per KV tile.
// ---------------------------------------------------------------------------
__global__ __launch_bounds__(256)
void attn_kernel(const _Float16* __restrict__ Q, const _Float16* __restrict__ K,
                 const _Float16* __restrict__ VT, _Float16* __restrict__ AO) {
    __shared__ _Float16 sK[2][128 * 72];      // K tile, pitch 72 f16 (144B)
    __shared__ _Float16 sV[2][64 * 136];      // VT tile, pitch 136 f16 (272B)
    __shared__ _Float16 sP[8 * 16 * 136];     // per-wave P buffer

    const int tid = threadIdx.x, w = tid >> 5, lane = tid & 31;
    const int ln = lane & 15, hf = lane >> 4;
    const int bh = blockIdx.y;
    const int qb = blockIdx.x * 128 + w * 16;
    const _Float16* Qh = Q  + (size_t)bh * SEQ * DHEAD;
    const _Float16* Kh = K  + (size_t)bh * SEQ * DHEAD;
    const _Float16* Vh = VT + (size_t)bh * DHEAD * SEQ;
    _Float16* Pw = sP + w * 16 * 136;

    // Q fragments, register-resident for all KV tiles
    v16h aq[2];
    {
        const _Float16* qr = Qh + (size_t)(qb + ln) * DHEAD;
#pragma unroll
        for (int ks = 0; ks < 2; ++ks)
            aq[ks] = cat8(*(const v8h*)(qr + ks * 32 + 8 * hf),
                          *(const v8h*)(qr + ks * 32 + 16 + 8 * hf));
    }

    v8f o[4];
    float mr[8], lrun[8];
#pragma unroll
    for (int i = 0; i < 4; ++i) o[i] = v8f{};
#pragma unroll
    for (int v = 0; v < 8; ++v) { mr[v] = -1e30f; lrun[v] = 0.0f; }

    const float kSc = 0.125f * 1.44269504088896340736f;  // 1/sqrt(64) * log2(e)

    // tile stagers
    auto stageK = [&](int kv, int buf) {
        const int r = tid >> 1, c = (tid & 1) * 32;
        const _Float16* gp = Kh + (size_t)(kv + r) * DHEAD + c;
        _Float16* sp = sK[buf] + r * 72 + c;
#pragma unroll
        for (int i = 0; i < 4; ++i) cp16(gp + i * 8, sp + i * 8);
    };
    auto stageV = [&](int kv, int buf) {
        const int r = tid >> 2, c = (tid & 3) * 32;
        const _Float16* gp = Vh + (size_t)r * SEQ + kv + c;
        _Float16* sp = sV[buf] + r * 136 + c;
#pragma unroll
        for (int i = 0; i < 4; ++i) cp16(gp + i * 8, sp + i * 8);
    };

    stageK(0, 0);
    stageV(0, 0);

    const int NKV = SEQ / 128;   // 16
    for (int it = 0; it < NKV; ++it) {
        const int buf = it & 1;
        async_fence();
        __syncthreads();
        if (it + 1 < NKV) { stageK((it + 1) * 128, buf ^ 1); stageV((it + 1) * 128, buf ^ 1); }

        // S = (Q K^T) * scale  -> 16 x 128 in 8 C-fragments
        v8f st[8];
#pragma unroll
        for (int nt = 0; nt < 8; ++nt) {
            v8f a = v8f{};
#pragma unroll
            for (int ks = 0; ks < 2; ++ks) {
                const _Float16* kr = sK[buf] + (nt * 16 + ln) * 72 + ks * 32;
                v16h bfr = cat8(*(const v8h*)(kr + 8 * hf), *(const v8h*)(kr + 16 + 8 * hf));
                a = wmma16(aq[ks], bfr, a);
            }
#pragma unroll
            for (int v = 0; v < 8; ++v) a[v] *= kSc;
            st[nt] = a;
        }

        // row max across n-tiles, then across 16 lanes of this half
        float mt[8];
#pragma unroll
        for (int v = 0; v < 8; ++v) {
            float m = st[0][v];
#pragma unroll
            for (int nt = 1; nt < 8; ++nt) m = fmaxf(m, st[nt][v]);
            mt[v] = m;
        }
#pragma unroll
        for (int off = 1; off < 16; off <<= 1)
#pragma unroll
            for (int v = 0; v < 8; ++v)
                mt[v] = fmaxf(mt[v], __shfl_xor(mt[v], off, 32));

        float corr[8];
#pragma unroll
        for (int v = 0; v < 8; ++v) {
            float mn = fmaxf(mr[v], mt[v]);
            corr[v]  = exp2f(mr[v] - mn);
            mr[v]    = mn;
        }

        // P = exp2(S - m) -> LDS (f16) for A-fragment reuse; row sums
        float rs[8];
#pragma unroll
        for (int v = 0; v < 8; ++v) rs[v] = 0.0f;
#pragma unroll
        for (int nt = 0; nt < 8; ++nt)
#pragma unroll
            for (int v = 0; v < 8; ++v) {
                float p = exp2f(st[nt][v] - mr[v]);
                Pw[(v + 8 * hf) * 136 + nt * 16 + ln] = (_Float16)p;
                rs[v] += p;
            }
#pragma unroll
        for (int off = 1; off < 16; off <<= 1)
#pragma unroll
            for (int v = 0; v < 8; ++v) rs[v] += __shfl_xor(rs[v], off, 32);
#pragma unroll
        for (int v = 0; v < 8; ++v) lrun[v] = lrun[v] * corr[v] + rs[v];

        // rescale running O
#pragma unroll
        for (int dt = 0; dt < 4; ++dt)
#pragma unroll
            for (int v = 0; v < 8; ++v) o[dt][v] *= corr[v];

        // O += P (16x128) x V (128x64)
#pragma unroll
        for (int ks = 0; ks < 4; ++ks) {
            const _Float16* pr = Pw + ln * 136 + ks * 32;
            v16h ap = cat8(*(const v8h*)(pr + 8 * hf),
                           *(const v8h*)(pr + 16 + 8 * hf));
#pragma unroll
            for (int dt = 0; dt < 4; ++dt) {
                const _Float16* vr = sV[buf] + (dt * 16 + ln) * 136 + ks * 32;
                v16h bfr = cat8(*(const v8h*)(vr + 8 * hf), *(const v8h*)(vr + 16 + 8 * hf));
                o[dt] = wmma16(ap, bfr, o[dt]);
            }
        }
    }

    // normalize + write AO[token][h*64+d] as f16
    const int b = bh >> 4, h = bh & 15;
    float inv[8];
#pragma unroll
    for (int v = 0; v < 8; ++v) inv[v] = 1.0f / lrun[v];
#pragma unroll
    for (int dt = 0; dt < 4; ++dt)
#pragma unroll
        for (int v = 0; v < 8; ++v) {
            const int s   = qb + v + 8 * hf;
            const int tr  = b * SEQ + s;
            const int col = h * DHEAD + dt * 16 + ln;
            AO[(size_t)tr * D_MODEL + col] = (_Float16)(o[dt][v] * inv[v]);
        }
}

// ---------------------------------------------------------------------------
// Kernel 5: output projection: out = AO @ Wo + bo  (f32 output)
// ---------------------------------------------------------------------------
__global__ __launch_bounds__(256)
void out_gemm(const _Float16* __restrict__ A, const _Float16* __restrict__ WT,
              const float* __restrict__ bo, float* __restrict__ out) {
    __shared__ _Float16 sA[2][128 * PITCH];
    __shared__ _Float16 sB[2][128 * PITCH];

    const int tid = threadIdx.x;
    const int w   = tid >> 5, lane = tid & 31;
    const int ln  = lane & 15, hf = lane >> 4;
    const int wm  = w >> 2, wn = w & 3;
    const int mb  = blockIdx.y * 128;
    const int nb  = blockIdx.x * 128;

    v8f acc[4][2];
#pragma unroll
    for (int i = 0; i < 4; ++i)
#pragma unroll
        for (int j = 0; j < 2; ++j) acc[i][j] = v8f{};

    const _Float16* gA = A  + (size_t)mb * D_MODEL;
    const _Float16* gB = WT + (size_t)nb * D_MODEL;

    stage_tile(gA, sA[0], tid);
    stage_tile(gB, sB[0], tid);

    const int NIT = D_MODEL / KT;
    for (int it = 0; it < NIT; ++it) {
        const int buf = it & 1;
        async_fence();
        __syncthreads();
        if (it + 1 < NIT) {
            stage_tile(gA + (it + 1) * KT, sA[buf ^ 1], tid);
            stage_tile(gB + (it + 1) * KT, sB[buf ^ 1], tid);
        }
#pragma unroll
        for (int ks = 0; ks < 2; ++ks) {
            v16h afr[4];
#pragma unroll
            for (int mt = 0; mt < 4; ++mt) {
                const _Float16* p = sA[buf] + (wm * 64 + mt * 16 + ln) * PITCH + ks * 32;
                afr[mt] = cat8(*(const v8h*)(p + 8 * hf), *(const v8h*)(p + 16 + 8 * hf));
            }
#pragma unroll
            for (int nt = 0; nt < 2; ++nt) {
                const _Float16* p = sB[buf] + (wn * 32 + nt * 16 + ln) * PITCH + ks * 32;
                v16h bfr = cat8(*(const v8h*)(p + 8 * hf), *(const v8h*)(p + 16 + 8 * hf));
#pragma unroll
                for (int mt = 0; mt < 4; ++mt)
                    acc[mt][nt] = wmma16(afr[mt], bfr, acc[mt][nt]);
            }
        }
    }

#pragma unroll
    for (int mt = 0; mt < 4; ++mt)
#pragma unroll
        for (int nt = 0; nt < 2; ++nt) {
            const int n = nb + wn * 32 + nt * 16 + ln;
            const float bn = bo[n];
#pragma unroll
            for (int v = 0; v < 8; ++v) {
                const int m = mb + wm * 64 + mt * 16 + v + 8 * hf;
                out[(size_t)m * D_MODEL + n] = acc[mt][nt][v] + bn;
            }
        }
}

// ---------------------------------------------------------------------------
extern "C" void kernel_launch(void* const* d_in, const int* in_sizes, int n_in,
                              void* d_out, int out_size, void* d_ws, size_t ws_size,
                              hipStream_t stream) {
    const float* x  = (const float*)d_in[0];
    const float* Wq = (const float*)d_in[1];
    const float* bq = (const float*)d_in[2];
    const float* Wk = (const float*)d_in[3];
    const float* bk = (const float*)d_in[4];
    const float* Wv = (const float*)d_in[5];
    const float* bv = (const float*)d_in[6];
    const float* Wo = (const float*)d_in[7];
    const float* bo = (const float*)d_in[8];
    float* out = (float*)d_out;

    const size_t NX = (size_t)NTOK * D_MODEL;      // 4M elements
    const size_t NW = (size_t)D_MODEL * D_MODEL;   // 1M elements
    _Float16* xf  = (_Float16*)d_ws;               // 4M
    _Float16* WT  = xf  + NX;                      // 4 x 1M (WqT,WkT,WvT,WoT)
    _Float16* Qb  = WT  + 4 * NW;                  // 4M
    _Float16* Kb  = Qb  + NX;                      // 4M
    _Float16* VTb = Kb  + NX;                      // 4M
    _Float16* AO  = VTb + NX;                      // 4M

    cvt_f32_to_f16<<<4096, 256, 0, stream>>>(x, xf, (int)NX);

    dim3 tg(32, 32);
    w_transpose_f16<<<tg, 256, 0, stream>>>(Wq, WT + 0 * NW);
    w_transpose_f16<<<tg, 256, 0, stream>>>(Wk, WT + 1 * NW);
    w_transpose_f16<<<tg, 256, 0, stream>>>(Wv, WT + 2 * NW);
    w_transpose_f16<<<tg, 256, 0, stream>>>(Wo, WT + 3 * NW);

    qkv_gemm<<<dim3(D_MODEL / 128, NTOK / 128, 3), 256, 0, stream>>>(
        xf, WT, bq, bk, bv, Qb, Kb, VTb);

    attn_kernel<<<dim3(SEQ / 128, BATCH * NHEAD), 256, 0, stream>>>(Qb, Kb, VTb, AO);

    out_gemm<<<dim3(D_MODEL / 128, NTOK / 128), 256, 0, stream>>>(AO, WT + 3 * NW, bo, out);
}
